// AEEncoder_31774168056077
// MI455X (gfx1250) — matrix operand order repaired and checked by reading, not compiled, not run on hardware
//
#include <hip/hip_runtime.h>
#include <stdint.h>

#define B_DIM   128
#define IN_F    25000
#define KP      25088        // IN_F padded to a multiple of 128 (K panel)
#define KPANEL  128          // K per LDS stage (4 WMMA steps)
#define NPANEL  (KP / KPANEL)
#define OUT_F   2048
#define N_EDGES 1000000
#define BN_EPS  1e-5f
#define SLOPE   0.01f

#define LDSROW  136          // padded row (ushorts): 272B = 68 dwords -> conflict-free

typedef __attribute__((ext_vector_type(16))) __bf16 v16bf;
typedef __attribute__((ext_vector_type(8)))  float  v8f;
typedef __attribute__((ext_vector_type(4)))  unsigned int u32x4;
typedef __attribute__((ext_vector_type(8)))  int  i32x8;
typedef __attribute__((ext_vector_type(4)))  int  i32x4;

#if __has_builtin(__builtin_amdgcn_tensor_load_to_lds)
#define HAVE_TDM 1
#else
#define HAVE_TDM 0
#endif

// ---------- bf16 helpers (round-to-nearest-even) ----------
static __device__ __forceinline__ unsigned short f2bf(float f) {
    union { float f; unsigned u; } cv; cv.f = f;
    unsigned u = cv.u;
    if ((u & 0x7fffffffu) > 0x7f800000u)            // NaN -> quiet NaN
        return (unsigned short)((u >> 16) | 0x40u);
    u += 0x7fffu + ((u >> 16) & 1u);                // RNE
    return (unsigned short)(u >> 16);
}
static __device__ __forceinline__ float bf2f(unsigned short h) {
    union { unsigned u; float f; } cv; cv.u = ((unsigned)h) << 16;
    return cv.f;
}

// ---------- 1. zero-fill dense W (grid-stride, 16B stores) ----------
__global__ void ae_zero_u4(uint4* __restrict__ p, long n4) {
    long i = (long)blockIdx.x * blockDim.x + threadIdx.x;
    long stride = (long)gridDim.x * blockDim.x;
    uint4 z; z.x = z.y = z.z = z.w = 0u;
    for (; i < n4; i += stride) p[i] = z;
}

// ---------- 2. scatter edges into dense bf16 W via 32-bit CAS ----------
__global__ void ae_scatter_w(const int* __restrict__ eo, const int* __restrict__ ei,
                             const float* __restrict__ w, unsigned* __restrict__ Wb) {
    int e = blockIdx.x * blockDim.x + threadIdx.x;
    if (e >= N_EDGES) return;
    int k = ei[e], n = eo[e];
    float val = w[e];
    unsigned idx  = (unsigned)k * OUT_F + (unsigned)n;
    unsigned word = idx >> 1;
    bool     hiH  = (idx & 1u) != 0;
    unsigned* p = Wb + word;
    unsigned old = *p, assumed;
    do {
        assumed = old;
        unsigned short h  = hiH ? (unsigned short)(assumed >> 16)
                                : (unsigned short)(assumed & 0xffffu);
        unsigned short nh = f2bf(bf2f(h) + val);
        unsigned nw = hiH ? ((assumed & 0x0000ffffu) | ((unsigned)nh << 16))
                          : ((assumed & 0xffff0000u) | (unsigned)nh);
        old = atomicCAS(p, assumed, nw);
    } while (old != assumed);
}

// ---------- 3. features f32 [B][IN_F] -> bf16 Xb [B][KP] (zero pad) ----------
__global__ void ae_cvt_x(const float* __restrict__ x, unsigned short* __restrict__ Xb) {
    long i = (long)blockIdx.x * blockDim.x + threadIdx.x;
    if (i >= (long)B_DIM * KP) return;
    int b = (int)(i / KP), k = (int)(i % KP);
    float v = (k < IN_F) ? x[(long)b * IN_F + k] : 0.0f;
    Xb[i] = f2bf(v);
}

// ---------- 4. GEMM: Y[128][2048] = Xb[128][KP] * Wb[KP][2048] ----------
// 128 blocks (16-col N tiles) x 8 waves (16-row M tiles). Double-buffered K
// panels: TDM DMA for the A (X) panel with LDS padding, one b128 + 8
// ds_store_b16 per thread for the transposed B (W) panel, 4 WMMAs per panel
// overlapped with the next panel's TDM + global loads.
__global__ __launch_bounds__(256)
void ae_gemm_wmma(const unsigned short* __restrict__ Xb,
                  const unsigned short* __restrict__ Wb,
                  float* __restrict__ Y) {
    __shared__ __align__(16) unsigned short Ash[2][B_DIM][LDSROW]; // 2 x 34,816 B
    __shared__ __align__(16) unsigned short Bst[2][16][LDSROW];    // 2 x  4,352 B

    const int nbase = blockIdx.x * 16;
    const int t     = threadIdx.x;
    const int wave  = t >> 5;
    const int lane  = t & 31;
    const int col   = lane & 15;
    const bool hi   = lane >= 16;
    const int mrow  = wave * 16 + col;

    // B staging role: thread covers K-row skk, N columns snh..snh+7.
    const int skk = t >> 1;
    const int snh = (t & 1) * 8;

    union U { uint4 q[2]; v16bf v; };
    union S { uint4 q; unsigned short h[8]; };
    static_assert(sizeof(U) == 32 && sizeof(S) == 16, "frag sizes");

    v8f acc = {0.f, 0.f, 0.f, 0.f, 0.f, 0.f, 0.f, 0.f};

#if HAVE_TDM
    // Loop-invariant TDM descriptor pieces (ISA 8.3/8.4), data_size=8B units.
    const unsigned lds_a[2] = {(unsigned)(uintptr_t)(void*)&Ash[0][0][0],
                               (unsigned)(uintptr_t)(void*)&Ash[1][0][0]};
    const unsigned long long xbase = (unsigned long long)(uintptr_t)Xb;
    const int g1_0 = (3 << 16)      // data_size = 8B
                   | (1 << 20)      // pad_enable
                   | (5 << 22)      // pad_interval: 64 dwords (256B row)
                   | (3 << 25);     // pad_amount:   4 dwords (16B) -> 272B stride
    const int g1_1 = (int)((KP / 4) & 0xffff) << 16;           // tensor_dim0 lo
    const int g1_2 = (int)(((KP / 4) >> 16) | (B_DIM << 16));  // dim0 hi | tensor_dim1 lo
    const int g1_3 = (int)((KPANEL / 4) << 16);                // dim1 hi | tile_dim0
    const int g1_4 = B_DIM;                                    // tile_dim1 | tile_dim2=0
    const int g1_5 = KP / 4;                                   // tensor_dim0_stride lo32
    const i32x8 g1 = {g1_0, g1_1, g1_2, g1_3, g1_4, g1_5, 0, 0};
    const i32x4 gz = {0, 0, 0, 0};

    auto issueA = [&](int kbn, unsigned ldsa) {
        unsigned long long ga = xbase + (unsigned long long)kbn * 2; // 256B aligned
        u32x4 g0 = {1u,                                    // count=1, user desc
                    ldsa,
                    (unsigned)ga,
                    (unsigned)((ga >> 32) & 0x1FFFFFFu) | (2u << 30)}; // type=2
#if defined(__clang_major__) && (__clang_major__ >= 23)
        const i32x8 gz8 = {0, 0, 0, 0, 0, 0, 0, 0};
        __builtin_amdgcn_tensor_load_to_lds(g0, g1, gz, gz, gz8, 0);
#else
        __builtin_amdgcn_tensor_load_to_lds(g0, g1, gz, gz, 0);
#endif
    };
#else
    auto copyA = [&](int kbn, int buf) {
        #pragma unroll
        for (int i = 0; i < 8; ++i) {
            int e = t + i * 256;           // 0..2047 16B chunks
            int r = e >> 4, c = e & 15;
            *(uint4*)&Ash[buf][r][c * 8] =
                *(const uint4*)(Xb + (long)r * KP + kbn + c * 8);
        }
    };
#endif

    // ---- prologue: stage panel 0 ----
#if HAVE_TDM
    if (wave == 0) issueA(0, lds_a[0]);
#else
    copyA(0, 0);
#endif
    {
        S s0; s0.q = *(const uint4*)(Wb + (long)skk * OUT_F + nbase + snh);
        #pragma unroll
        for (int j = 0; j < 8; ++j) Bst[0][snh + j][skk] = s0.h[j];
    }

    for (int p = 0; p < NPANEL; ++p) {
        const int kb  = p * KPANEL;
        const int cur = p & 1, nxt = cur ^ 1;
        const bool more = (p + 1) < NPANEL;

#if HAVE_TDM
        if (wave == 0) __builtin_amdgcn_s_wait_tensorcnt(0); // TDM(p) complete
#endif
        __syncthreads();                                     // panel p visible

        // Kick off panel p+1 (overlaps with this panel's WMMAs).
#if HAVE_TDM
        if (more && wave == 0) issueA(kb + KPANEL, lds_a[nxt]);
#else
        if (more) copyA(kb + KPANEL, nxt);
#endif
        S sb;
        if (more)
            sb.q = *(const uint4*)(Wb + (long)(kb + KPANEL + skk) * OUT_F + nbase + snh);

        // 4 WMMA steps over the current 128-K panel.
        #pragma unroll
        for (int s = 0; s < 4; ++s) {
            U a, b;
            const unsigned short* ap = &Ash[cur][mrow][32 * s + (hi ? 8 : 0)];
            a.q[0] = *(const uint4*)(ap);
            a.q[1] = *(const uint4*)(ap + 16);
            const unsigned short* bp = &Bst[cur][col][32 * s + (hi ? 8 : 0)];
            b.q[0] = *(const uint4*)(bp);
            b.q[1] = *(const uint4*)(bp + 16);
            acc = __builtin_amdgcn_wmma_f32_16x16x32_bf16(
                      false, a.v, false, b.v, (short)0, acc, false, false);
        }

        // Commit staged B data for panel p+1 (before the next top barrier).
        if (more) {
            #pragma unroll
            for (int j = 0; j < 8; ++j) Bst[nxt][snh + j][skk] = sb.h[j];
        }
    }

    // C layout: VGPR r = row (wave*16 + r) lanes 0-15, +8 for lanes 16-31.
    const int mbase = wave * 16 + (hi ? 8 : 0);
    #pragma unroll
    for (int r = 0; r < 8; ++r)
        Y[(long)(mbase + r) * OUT_F + nbase + col] = acc[r];
}

// ---------- 5. BatchNorm(affine=False, batch stats) + LeakyReLU, in place ----
__global__ void ae_bn_lrelu(float* __restrict__ Y) {
    int o = blockIdx.x * blockDim.x + threadIdx.x;
    if (o >= OUT_F) return;
    float s = 0.f;
    for (int b = 0; b < B_DIM; ++b) s += Y[(long)b * OUT_F + o];
    float mean = s * (1.0f / B_DIM);
    float vs = 0.f;
    for (int b = 0; b < B_DIM; ++b) {
        float d = Y[(long)b * OUT_F + o] - mean;
        vs += d * d;
    }
    float inv = rsqrtf(vs * (1.0f / B_DIM) + BN_EPS);
    for (int b = 0; b < B_DIM; ++b) {
        float v = (Y[(long)b * OUT_F + o] - mean) * inv;
        Y[(long)b * OUT_F + o] = (v >= 0.f) ? v : SLOPE * v;
    }
}

extern "C" void kernel_launch(void* const* d_in, const int* in_sizes, int n_in,
                              void* d_out, int out_size, void* d_ws, size_t ws_size,
                              hipStream_t stream) {
    const float* features = (const float*)d_in[0];
    const float* weight   = (const float*)d_in[1];
    // d_in[2] = bias: provably a no-op (BatchNorm subtracts the batch mean).
    const int* edge_out = (const int*)d_in[3];
    const int* edge_in  = (const int*)d_in[4];
    float* Y = (float*)d_out;

    // Workspace: dense bf16 W [KP][OUT_F] (102.8 MB, L2-resident on 192MB L2),
    // then bf16 X [B][KP] (6.4 MB).
    unsigned short* Wb = (unsigned short*)d_ws;
    unsigned short* Xb = (unsigned short*)((char*)d_ws + (size_t)KP * OUT_F * 2);

    long n4 = ((long)KP * OUT_F * 2) / 16;
    ae_zero_u4<<<4096, 256, 0, stream>>>((uint4*)Wb, n4);

    ae_scatter_w<<<(N_EDGES + 255) / 256, 256, 0, stream>>>(
        edge_out, edge_in, weight, (unsigned*)Wb);

    long nx = (long)B_DIM * KP;
    ae_cvt_x<<<(int)((nx + 255) / 256), 256, 0, stream>>>(features, Xb);

    ae_gemm_wmma<<<OUT_F / 16, 256, 0, stream>>>(Xb, Wb, Y);

    ae_bn_lrelu<<<OUT_F / 256, 256, 0, stream>>>(Y);
}
